// MultiHeadLinearAttention_26431228739641
// MI455X (gfx1250) — matrix operand order repaired and checked by reading, not compiled, not run on hardware
//
#include <hip/hip_runtime.h>
#include <hip/hip_bf16.h>
#include <stdint.h>

// ---------------------------------------------------------------------------
// MI455X (gfx1250, wave32) multi-head linear attention forward.
// GEMMs via v_wmma_f32_16x16x32_bf16 with TDM (tensor_load_to_lds) double-
// buffered LDS staging; operands stored K-contiguous so fragment loads are
// b128 pairs matching the CDNA5 wave32 VGPR layouts.
// ---------------------------------------------------------------------------

#define D_MODEL 1024
#define NHEADS  16
#define HDIM    64
#define BATCH   4
#define SEQ     8192
#define MTOT    (BATCH * SEQ)       // 32768 tokens
#define BH      (BATCH * NHEADS)    // 64

typedef __attribute__((ext_vector_type(16))) __bf16 v16bf;
typedef __attribute__((ext_vector_type(8)))  float  v8f;
typedef __attribute__((ext_vector_type(4)))  unsigned int v4u;
typedef __attribute__((ext_vector_type(4)))  int v4i;
typedef __attribute__((ext_vector_type(8)))  int v8i;

union Frag { v16bf v; uint4 q[2]; };

__device__ __forceinline__ float bf2f(uint16_t u) {
  union { uint32_t u; float f; } c; c.u = ((uint32_t)u) << 16; return c.f;
}
__device__ __forceinline__ uint16_t f2bf(float f) {
  union { float f; uint32_t u; } c; c.f = f;
  uint32_t r = c.u + 0x7FFFu + ((c.u >> 16) & 1u);   // round-to-nearest-even
  return (uint16_t)(r >> 16);
}

// ---------------- Tensor Data Mover: 2D bf16 tile -> LDS --------------------
// Builds D# groups per CDNA5 ISA ch.8 and issues TENSOR_LOAD_TO_LDS.
// (6-arg builtin form: g0, g1, g2, g3, g4, cpol.)
__device__ __forceinline__ void tdm_load_tile_2d(
    unsigned lds_off,            // byte offset into LDS
    const uint16_t* gsrc,        // global address of tile start
    unsigned tensor_d0,          // tensor dim0 length (elements)
    unsigned tensor_d1,          // tensor dim1 length (rows)
    unsigned tile_d0,            // tile width  (elements, contiguous)
    unsigned tile_d1,            // tile height (rows)
    unsigned row_stride)         // dim0 stride (elements)
{
  unsigned long long ga = (unsigned long long)(uintptr_t)gsrc;
  v4u g0;
  g0.x = 1u;                                            // count=1 (valid), user mode
  g0.y = lds_off;                                       // lds_addr [63:32]
  g0.z = (unsigned)(ga & 0xFFFFFFFFu);                  // global_addr [95:64]
  g0.w = (unsigned)((ga >> 32) & 0x01FFFFFFu) | (2u << 30); // global_addr hi | type=2
  v8i g1;
  g1[0] = (int)(1u << 16);                              // data_size=1 (2B), wg_mask=0
  g1[1] = (int)((tensor_d0 & 0xFFFFu) << 16);           // tensor_dim0[15:0]
  g1[2] = (int)(((tensor_d0 >> 16) & 0xFFFFu) |
                ((tensor_d1 & 0xFFFFu) << 16));         // dim0 hi | dim1 lo
  g1[3] = (int)(((tensor_d1 >> 16) & 0xFFFFu) |
                ((tile_d0 & 0xFFFFu) << 16));           // dim1 hi | tile_dim0
  g1[4] = (int)(tile_d1 & 0xFFFFu);                     // tile_dim1 (tile_dim2=0)
  g1[5] = (int)row_stride;                              // tensor_dim0_stride[31:0]
  g1[6] = 0;                                            // stride hi, dim1_stride lo
  g1[7] = 0;
  v4i z4 = {0, 0, 0, 0};
  v8i z8 = {0, 0, 0, 0, 0, 0, 0, 0};
  __builtin_amdgcn_tensor_load_to_lds(g0, g1, z4, z4, z8, 0);
}

__device__ __forceinline__ unsigned lds_offset_of(const void* p) {
  return (unsigned)(uintptr_t)p;   // generic LDS pointer: low 32 bits = LDS offset
}

// ---------------- conversion kernels ----------------------------------------

__global__ void cvt_f32_bf16(const float* __restrict__ in,
                             uint16_t* __restrict__ out, int n) {
  int i = blockIdx.x * 256 + threadIdx.x;
  if (i < n) out[i] = f2bf(in[i]);
}

// W (K=1024 x N=1024 row-major fp32)  ->  WT (N x K row-major bf16)
__global__ void cvt_transpose_1024(const float* __restrict__ in,
                                   uint16_t* __restrict__ out) {
  int idx = blockIdx.x * 256 + threadIdx.x;   // 1024*1024 total
  int n = idx & 1023;
  int k = idx >> 10;
  out[(size_t)n * 1024 + k] = f2bf(in[(size_t)k * 1024 + n]);
}

__global__ void zero_f32(float* __restrict__ p, int n) {
  int i = blockIdx.x * 256 + threadIdx.x;
  if (i < n) p[i] = 0.0f;
}

// ---------------- main WMMA GEMM (TDM double-buffered) ----------------------
// C(M=MTOT, N=1024) = A_bf16 @ BT_bf16^T + bias
// MODE 0: store bf16 head-major           (b,h,n,dh)
// MODE 1: store fp32 row-major to d_out   (token, 1024)
// MODE 2: store bf16 head-transposed      (b,h,dh,n)
template <int MODE>
__global__ __launch_bounds__(256) void gemm_bf16(
    const uint16_t* __restrict__ A,    // MTOT x 1024, row-major, K-contiguous
    const uint16_t* __restrict__ BT,   // 1024 x 1024, BT[n][k], K-contiguous
    const float*    __restrict__ bias, // 1024
    void*           __restrict__ out) {
  __shared__ __align__(16) uint16_t lA[2][128 * 32];
  __shared__ __align__(16) uint16_t lB[2][128 * 32];

  const int tid  = threadIdx.x;
  const int lane = tid & 31;
  const int wave = tid >> 5;
  const int wm   = wave >> 1;      // 0..3, M sub-tile of 32
  const int wn   = wave & 1;       // 0..1, N sub-tile of 64

  const int m0 = blockIdx.x * 128;
  const int n0 = blockIdx.y * 128;

  v8f acc[2][4] = {};

  // per-lane fragment offsets (CDNA5 wave32 16-bit layouts)
  const int aM  = lane & 15;            // A: row M
  const int aKb = (lane >> 4) * 8;      // A: chunks {aKb..+7, aKb+16..+23}
  const int bN  = lane & 15;            // B: column N
  const int bKb = (lane >> 4) * 8;      // B: chunks {bKb..+7, bKb+16..+23}

  const bool issuer = (wave == 0);

  // prologue: DMA the first K-slab into buffer 0
  if (issuer) {
    tdm_load_tile_2d(lds_offset_of(&lA[0][0]), A  + (size_t)m0 * 1024,
                     1024, MTOT, 32, 128, 1024);
    tdm_load_tile_2d(lds_offset_of(&lB[0][0]), BT + (size_t)n0 * 1024,
                     1024, 1024, 32, 128, 1024);
  }

  for (int kt = 0; kt < 32; ++kt) {
    const int cur = kt & 1;
    const int nxt = cur ^ 1;
    if (issuer) {
      if (kt + 1 < 32) {
        const int k1 = (kt + 1) * 32;
        tdm_load_tile_2d(lds_offset_of(&lA[nxt][0]), A  + (size_t)m0 * 1024 + k1,
                         1024, MTOT, 32, 128, 1024);
        tdm_load_tile_2d(lds_offset_of(&lB[nxt][0]), BT + (size_t)n0 * 1024 + k1,
                         1024, 1024, 32, 128, 1024);
        __builtin_amdgcn_s_wait_tensorcnt(2);  // current buffer's pair complete
      } else {
        __builtin_amdgcn_s_wait_tensorcnt(0);
      }
    }
    __syncthreads();   // current buffer visible to all waves

    Frag a[2], b[4];
#pragma unroll
    for (int i = 0; i < 2; ++i) {
      const uint16_t* base = &lA[cur][(wm * 32 + i * 16 + aM) * 32];
      a[i].q[0] = *(const uint4*)(base + aKb);
      a[i].q[1] = *(const uint4*)(base + aKb + 16);
    }
#pragma unroll
    for (int j = 0; j < 4; ++j) {
      const uint16_t* base = &lB[cur][(wn * 64 + j * 16 + bN) * 32 + bKb];
      b[j].q[0] = ((const uint4*)base)[0];
      b[j].q[1] = ((const uint4*)base)[2];
    }
#pragma unroll
    for (int i = 0; i < 2; ++i)
#pragma unroll
      for (int j = 0; j < 4; ++j)
        acc[i][j] = __builtin_amdgcn_wmma_f32_16x16x32_bf16(
            false, a[i].v, false, b[j].v, (short)0, acc[i][j], false, false);
    __syncthreads();   // reads done; buffer may be overwritten next iter
  }

  // epilogue. C/D layout: lanes 0-15 -> M=r, lanes 16-31 -> M=r+8; N = lane%16
  const int mOffHi = (lane >> 4) << 3;
#pragma unroll
  for (int j = 0; j < 4; ++j) {
    const int col = n0 + wn * 64 + j * 16 + (lane & 15);
    const float bv = bias[col];
#pragma unroll
    for (int i = 0; i < 2; ++i) {
      const int rowBase = m0 + wm * 32 + i * 16 + mOffHi;
#pragma unroll
      for (int r = 0; r < 8; ++r) {
        const int row = rowBase + r;
        const float val = acc[i][j][r] + bv;
        if (MODE == 1) {
          ((float*)out)[(size_t)row * 1024 + col] = val;
        } else {
          const int b_ = row >> 13, n_ = row & 8191;
          const int h_ = col >> 6,  d_ = col & 63;
          if (MODE == 0)
            ((uint16_t*)out)[(((size_t)(b_ * 16 + h_) * SEQ + n_) << 6) + d_] = f2bf(val);
          else  // MODE 2: (b,h,dh,n)
            ((uint16_t*)out)[((size_t)(b_ * 16 + h_) * 64 + d_) * SEQ + n_] = f2bf(val);
        }
      }
    }
  }
}

// ---------------- softmax over dh=64, one wave32 per row --------------------

__global__ void softmax64_inplace(uint16_t* __restrict__ q, long nrows) {
  long row = (long)blockIdx.x * 8 + (threadIdx.x >> 5);
  if (row >= nrows) return;
  int lane = threadIdx.x & 31;
  uint16_t* p = q + row * 64 + lane * 2;
  uint32_t two = *(const uint32_t*)p;
  float x0 = bf2f((uint16_t)two), x1 = bf2f((uint16_t)(two >> 16));
  float m = fmaxf(x0, x1);
  for (int o = 16; o > 0; o >>= 1) m = fmaxf(m, __shfl_xor(m, o, 32));
  float e0 = __expf(x0 - m), e1 = __expf(x1 - m);
  float s = e0 + e1;
  for (int o = 16; o > 0; o >>= 1) s += __shfl_xor(s, o, 32);
  float inv = 1.0f / s;
  *(uint32_t*)p = (uint32_t)f2bf(e0 * inv) | ((uint32_t)f2bf(e1 * inv) << 16);
}

// softmax k rows and scatter into kT (b,h,dh,n) so the context GEMM's A
// operand is K(n)-contiguous.
__global__ void softmax64_k_transpose(const uint16_t* __restrict__ k,
                                      uint16_t* __restrict__ kT, long nrows) {
  long row = (long)blockIdx.x * 8 + (threadIdx.x >> 5);  // row = bh*SEQ + n
  if (row >= nrows) return;
  int lane = threadIdx.x & 31;
  uint32_t two = *(const uint32_t*)(k + row * 64 + lane * 2);
  float x0 = bf2f((uint16_t)two), x1 = bf2f((uint16_t)(two >> 16));
  float m = fmaxf(x0, x1);
  for (int o = 16; o > 0; o >>= 1) m = fmaxf(m, __shfl_xor(m, o, 32));
  float e0 = __expf(x0 - m), e1 = __expf(x1 - m);
  float s = e0 + e1;
  for (int o = 16; o > 0; o >>= 1) s += __shfl_xor(s, o, 32);
  float inv = 1.0f / s;
  long bh = row >> 13, n = row & 8191;
  kT[(bh * 64 + lane * 2 + 0) * SEQ + n] = f2bf(e0 * inv);
  kT[(bh * 64 + lane * 2 + 1) * SEQ + n] = f2bf(e1 * inv);
}

// ksum[bh*64+dd] = sum_n kT[bh,dd,n] — one wave per (bh,dd) row
__global__ void ksum_kernel(const uint16_t* __restrict__ kT,
                            float* __restrict__ ksum) {
  int idx = blockIdx.x * 8 + (threadIdx.x >> 5);   // 0..4095
  int lane = threadIdx.x & 31;
  const uint16_t* p = kT + (size_t)idx * SEQ;
  float s = 0.f;
  for (int i = lane * 2; i < SEQ; i += 64) {
    uint32_t two = *(const uint32_t*)(p + i);
    s += bf2f((uint16_t)two) + bf2f((uint16_t)(two >> 16));
  }
  for (int o = 16; o > 0; o >>= 1) s += __shfl_xor(s, o, 32);
  if (lane == 0) ksum[idx] = s;
}

// d_inv[bh*SEQ+n] = 1 / sum_dd( q * ksum )
__global__ void dinv_kernel(const uint16_t* __restrict__ q,
                            const float* __restrict__ ksum,
                            float* __restrict__ dinv, long nrows) {
  long row = (long)blockIdx.x * 8 + (threadIdx.x >> 5);
  if (row >= nrows) return;
  int lane = threadIdx.x & 31;
  long bh = row >> 13;
  uint32_t two = *(const uint32_t*)(q + row * 64 + lane * 2);
  const float* ks = ksum + bh * 64 + lane * 2;
  float s = bf2f((uint16_t)two) * ks[0] + bf2f((uint16_t)(two >> 16)) * ks[1];
  for (int o = 16; o > 0; o >>= 1) s += __shfl_xor(s, o, 32);
  if (lane == 0) dinv[row] = 1.0f / s;
}

// ---------------- context: ctx[d][e] = sum_n k[n,d] v[n,e]  (K = 8192) ------
// Split the n-reduction into 8 chunks for occupancy; accumulate fp32 partials
// with global atomics into ctxAcc laid out transposed (bh, e, d).
__global__ __launch_bounds__(128) void context_partial(
    const uint16_t* __restrict__ kT,   // (bh, dh, n)
    const uint16_t* __restrict__ vT,   // (bh, dh, n)
    float* __restrict__ ctxAcc) {      // (bh, e, d) fp32, pre-zeroed
  const int bh    = blockIdx.x >> 3;   // 64
  const int chunk = blockIdx.x & 7;    // 8 chunks of 1024 over n
  const int wave  = threadIdx.x >> 5;  // d-tile of 16
  const int lane  = threadIdx.x & 31;
  const uint16_t* kbase = kT + (size_t)bh * 64 * SEQ;
  const uint16_t* vbase = vT + (size_t)bh * 64 * SEQ;

  const int aM = lane & 15, aKb = (lane >> 4) * 8;
  const int bN = lane & 15, bKb = (lane >> 4) * 8;
  const uint16_t* arow = kbase + (size_t)(wave * 16 + aM) * SEQ;
  const int nBeg = chunk * 1024, nEnd = nBeg + 1024;

  v8f acc[4] = {};
  for (int nt = nBeg; nt < nEnd; nt += 32) {
    Frag a;
    a.q[0] = *(const uint4*)(arow + nt + aKb);
    a.q[1] = *(const uint4*)(arow + nt + aKb + 16);
#pragma unroll
    for (int j = 0; j < 4; ++j) {
      Frag b;
      const uint16_t* br = vbase + (size_t)(j * 16 + bN) * SEQ + nt + bKb;
      b.q[0] = ((const uint4*)br)[0];
      b.q[1] = ((const uint4*)br)[2];
      acc[j] = __builtin_amdgcn_wmma_f32_16x16x32_bf16(
          false, a.v, false, b.v, (short)0, acc[j], false, false);
    }
  }
#pragma unroll
  for (int j = 0; j < 4; ++j) {
    const int e = j * 16 + (lane & 15);
#pragma unroll
    for (int r = 0; r < 8; ++r) {
      const int d = wave * 16 + r + ((lane >> 4) << 3);
      atomicAdd(&ctxAcc[((size_t)bh * 64 + e) * 64 + d], acc[j][r]);
    }
  }
}

__global__ void cvt_ctx_bf16(const float* __restrict__ ctxAcc,
                             uint16_t* __restrict__ ctxT, int n) {
  int i = blockIdx.x * 256 + threadIdx.x;
  if (i < n) ctxT[i] = f2bf(ctxAcc[i]);
}

// ---------------- y = (q @ ctx) * d_inv + q, scattered to (b,n,d_model) -----
__global__ __launch_bounds__(128) void attn_y_kernel(
    const uint16_t* __restrict__ qb,    // (bh, n, dh)
    const uint16_t* __restrict__ ctxT,  // (bh, e, d)
    const float*    __restrict__ dinv,  // (bh, n)
    uint16_t*       __restrict__ yb) {  // (b, n, d_model)
  const int blk  = blockIdx.x;          // 64 * (SEQ/64) = 8192
  const int bh   = blk >> 7;
  const int tblk = blk & 127;
  const int wave = threadIdx.x >> 5;
  const int lane = threadIdx.x & 31;
  const int tok0 = tblk * 64 + wave * 16;

  const int aM = lane & 15, aKb = (lane >> 4) * 8;
  const int bN = lane & 15, bKb = (lane >> 4) * 8;
  const uint16_t* qrow = qb + ((size_t)bh * SEQ + tok0 + aM) * 64;

  v8f acc[4] = {};
#pragma unroll
  for (int kt = 0; kt < 2; ++kt) {
    Frag a;
    a.q[0] = *(const uint4*)(qrow + kt * 32 + aKb);
    a.q[1] = *(const uint4*)(qrow + kt * 32 + aKb + 16);
#pragma unroll
    for (int j = 0; j < 4; ++j) {
      Frag b;
      const uint16_t* br = ctxT + ((size_t)bh * 64 + j * 16 + bN) * 64 + kt * 32 + bKb;
      b.q[0] = ((const uint4*)br)[0];
      b.q[1] = ((const uint4*)br)[2];
      acc[j] = __builtin_amdgcn_wmma_f32_16x16x32_bf16(
          false, a.v, false, b.v, (short)0, acc[j], false, false);
    }
  }
  const int b_ = bh >> 4, h_ = bh & 15;
#pragma unroll
  for (int j = 0; j < 4; ++j) {
    const int e = j * 16 + (lane & 15);
#pragma unroll
    for (int r = 0; r < 8; ++r) {
      const int n = tok0 + r + ((lane >> 4) << 3);
      const float di = dinv[(size_t)bh * SEQ + n];
      const float qv = bf2f(qb[((size_t)bh * SEQ + n) * 64 + e]);
      yb[((size_t)b_ * SEQ + n) * 1024 + h_ * 64 + e] = f2bf(acc[j][r] * di + qv);
    }
  }
}

// ---------------------------------------------------------------------------

extern "C" void kernel_launch(void* const* d_in, const int* in_sizes, int n_in,
                              void* d_out, int out_size, void* d_ws, size_t ws_size,
                              hipStream_t stream) {
  const float* x  = (const float*)d_in[0];
  const float* Wq = (const float*)d_in[1];
  const float* bq = (const float*)d_in[2];
  const float* Wk = (const float*)d_in[3];
  const float* bk = (const float*)d_in[4];
  const float* Wv = (const float*)d_in[5];
  const float* bv = (const float*)d_in[6];
  const float* Wo = (const float*)d_in[7];
  const float* bo = (const float*)d_in[8];

  // workspace layout (256B aligned slabs)
  auto align256 = [](size_t v) { return (v + 255) & ~(size_t)255; };
  char* ws = (char*)d_ws;
  size_t off = 0;
  auto take = [&](size_t bytes) { char* p = ws + off; off = align256(off + bytes); return p; };

  const size_t XE = (size_t)MTOT * D_MODEL;       // 33.5M elems
  uint16_t* xb   = (uint16_t*)take(XE * 2);
  uint16_t* WqT  = (uint16_t*)take((size_t)D_MODEL * D_MODEL * 2);
  uint16_t* WkT  = (uint16_t*)take((size_t)D_MODEL * D_MODEL * 2);
  uint16_t* WvT  = (uint16_t*)take((size_t)D_MODEL * D_MODEL * 2);
  uint16_t* WoT  = (uint16_t*)take((size_t)D_MODEL * D_MODEL * 2);
  uint16_t* qb   = (uint16_t*)take(XE * 2);       // (b,h,n,dh)
  uint16_t* kbuf = (uint16_t*)take(XE * 2);       // (b,h,n,dh), reused as yb
  uint16_t* kT   = (uint16_t*)take(XE * 2);       // (b,h,dh,n)
  uint16_t* vT   = (uint16_t*)take(XE * 2);       // (b,h,dh,n)
  float*    ksum = (float*)take((size_t)BH * HDIM * 4);
  float*    dinv = (float*)take((size_t)BH * SEQ * 4);
  float*    ctxA = (float*)take((size_t)BH * HDIM * HDIM * 4);
  uint16_t* ctxT = (uint16_t*)take((size_t)BH * HDIM * HDIM * 2);
  uint16_t* yb   = kbuf;                          // k (n-major) dead by then

  const long NROWS = (long)BH * SEQ;              // 524288 head-rows
  const int  CTXN  = BH * HDIM * HDIM;            // 262144

  // 1) downcast activations + transposed-downcast weights
  cvt_f32_bf16<<<(int)((XE + 255) / 256), 256, 0, stream>>>(x, xb, (int)XE);
  cvt_transpose_1024<<<4096, 256, 0, stream>>>(Wq, WqT);
  cvt_transpose_1024<<<4096, 256, 0, stream>>>(Wk, WkT);
  cvt_transpose_1024<<<4096, 256, 0, stream>>>(Wv, WvT);
  cvt_transpose_1024<<<4096, 256, 0, stream>>>(Wo, WoT);
  zero_f32<<<(CTXN + 255) / 256, 256, 0, stream>>>(ctxA, CTXN);

  // 2) Q/K/V projections (WMMA bf16, TDM double-buffered)
  dim3 gg(MTOT / 128, D_MODEL / 128);
  gemm_bf16<0><<<gg, 256, 0, stream>>>(xb, WqT, bq, (void*)qb);
  gemm_bf16<0><<<gg, 256, 0, stream>>>(xb, WkT, bk, (void*)kbuf);
  gemm_bf16<2><<<gg, 256, 0, stream>>>(xb, WvT, bv, (void*)vT);

  // 3) softmax over dh; k is transposed to (b,h,dh,n) on the way out
  softmax64_inplace<<<(int)(NROWS / 8), 256, 0, stream>>>(qb, NROWS);
  softmax64_k_transpose<<<(int)(NROWS / 8), 256, 0, stream>>>(kbuf, kT, NROWS);

  // 4) k_sum and d_inv
  ksum_kernel<<<BH * HDIM / 8, 256, 0, stream>>>(kT, ksum);
  dinv_kernel<<<(int)(NROWS / 8), 256, 0, stream>>>(qb, ksum, dinv, NROWS);

  // 5) context = k^T v (per head, K=8192, 8-way split + fp32 atomics)
  context_partial<<<BH * 8, 128, 0, stream>>>(kT, vT, ctxA);
  cvt_ctx_bf16<<<(CTXN + 255) / 256, 256, 0, stream>>>(ctxA, ctxT, CTXN);

  // 6) y = q@ctx * d_inv + q, then output projection -> fp32 d_out
  attn_y_kernel<<<BH * (SEQ / 64), 128, 0, stream>>>(qb, ctxT, dinv, yb);
  gemm_bf16<1><<<gg, 256, 0, stream>>>(yb, WoT, bo, d_out);
}
